// NormAttention_9079560863963
// MI455X (gfx1250) — compile-verified, matrix operand order checked by reading, not run on hardware
//
#include <hip/hip_runtime.h>

// ---------------------------------------------------------------------------
// Problem dims (fixed by the reference): B=2, N=2048, E=1024, H=8, D=128
// Tokens T = B*N = 4096, HID = 1024.
// ---------------------------------------------------------------------------
#define T_TOK   4096
#define E_DIM   1024
#define HID_DIM 1024
#define N_SEQ   2048
#define H_HEADS 8
#define D_HEAD  128
#define EPS_F   1e-8f

typedef __attribute__((ext_vector_type(16))) __bf16 v16bf;
typedef __attribute__((ext_vector_type(8)))  float  v8f;
typedef unsigned int   u32;
typedef unsigned short u16;

union Frag16 {            // 16 bf16 per lane = 32 bytes = 8 VGPRs
    v16bf bf;
    uint4 u[2];
};

__device__ __forceinline__ u16 f2bf(float f) {
    u32 u = __float_as_uint(f);
    u32 r = 0x7FFFu + ((u >> 16) & 1u);   // round-to-nearest-even
    u += r;
    return (u16)(u >> 16);
}

__device__ __forceinline__ float silu_f(float v) {
    return v / (1.0f + __expf(-v));
}
__device__ __forceinline__ float elu_f(float v) {
    return v > 0.0f ? v : (__expf(v) - 1.0f);
}

// Load a 16-bit WMMA A/B fragment for one lane. `p` points at this lane's
// row (A: row M=lane%16 of A; B: row N=lane%16 of the K-contiguous operand)
// offset to the current K position. Per ISA layout: lanes 0-15 hold K+{0..7},
// K+{16..23}; lanes 16-31 hold K+{8..15}, K+{24..31}.
__device__ __forceinline__ v16bf load_frag(const u16* p, int lane) {
    const int k0 = (lane & 16) ? 8 : 0;
    Frag16 f;
    f.u[0] = *(const uint4*)(p + k0);
    f.u[1] = *(const uint4*)(p + k0 + 16);
    return f.bf;
}

// ---------------------------------------------------------------------------
// 0) fp32 -> bf16 conversion (vectorized by 4)
// ---------------------------------------------------------------------------
__global__ void cvt_bf16_kernel(const float* __restrict__ src,
                                u16* __restrict__ dst, int n4) {
    int i = blockIdx.x * blockDim.x + threadIdx.x;
    if (i < n4) {
        float4 v = ((const float4*)src)[i];
        u32 lo = (u32)f2bf(v.x) | ((u32)f2bf(v.y) << 16);
        u32 hi = (u32)f2bf(v.z) | ((u32)f2bf(v.w) << 16);
        ((uint2*)dst)[i] = make_uint2(lo, hi);
    }
}

// ---------------------------------------------------------------------------
// 1) Fused Q/K/V/U projection: out = act(x @ W^T + b)
//    One wave computes a 32(M) x 64(N) tile. out_id: 0=Q(elu,bf16),
//    1=K(elu,bf16), 2=V(silu,bf16 transposed Vt[d][tok]), 3=U(silu,f32).
// ---------------------------------------------------------------------------
__global__ __launch_bounds__(256)
void proj_kernel(const u16* __restrict__ xh, const u16* __restrict__ W4h,
                 const float* __restrict__ bq, const float* __restrict__ bk,
                 const float* __restrict__ bv, const float* __restrict__ bu,
                 u16* __restrict__ Qh, u16* __restrict__ Kh,
                 u16* __restrict__ Vt, float* __restrict__ Uf) {
    const int lane = threadIdx.x & 31;
    const int wid  = (blockIdx.x * blockDim.x + threadIdx.x) >> 5;
    const int out_id = wid >> 11;          // 2048 tiles per output
    const int t  = wid & 2047;
    const int tm = (t >> 4) << 5;          // 128 M-tiles of 32
    const int tn = (t & 15) << 6;          // 16  N-tiles of 64
    const int lr = lane & 15;
    const int rbase = (lane >> 4) << 3;    // 0 or 8 (C-layout row base)

    const u16* W = W4h + (size_t)out_id * (HID_DIM * E_DIM);
    const float* bias = (out_id == 0) ? bq : (out_id == 1) ? bk
                      : (out_id == 2) ? bv : bu;

    const u16* arow0 = xh + (size_t)(tm + lr)      * E_DIM;
    const u16* arow1 = xh + (size_t)(tm + 16 + lr) * E_DIM;
    const u16* brow[4];
#pragma unroll
    for (int s = 0; s < 4; ++s)
        brow[s] = W + (size_t)(tn + s * 16 + lr) * E_DIM;

    v8f acc[2][4] = {};
    for (int kk = 0; kk < E_DIM; kk += 32) {
        v16bf a0 = load_frag(arow0 + kk, lane);
        v16bf a1 = load_frag(arow1 + kk, lane);
#pragma unroll
        for (int s = 0; s < 4; ++s) {
            v16bf b = load_frag(brow[s] + kk, lane);
            acc[0][s] = __builtin_amdgcn_wmma_f32_16x16x32_bf16(
                false, a0, false, b, (short)0, acc[0][s], false, false);
            acc[1][s] = __builtin_amdgcn_wmma_f32_16x16x32_bf16(
                false, a1, false, b, (short)0, acc[1][s], false, false);
        }
    }

#pragma unroll
    for (int sm = 0; sm < 2; ++sm) {
#pragma unroll
        for (int sn = 0; sn < 4; ++sn) {
            const int col  = tn + sn * 16 + lr;
            const float bc = bias[col];
            const int row0 = tm + sm * 16 + rbase;
            if (out_id <= 1) {                       // ELU -> bf16, row-major
                u16* dst = (out_id == 0) ? Qh : Kh;
#pragma unroll
                for (int j = 0; j < 8; ++j) {
                    float v = elu_f(acc[sm][sn][j] + bc);
                    dst[(size_t)(row0 + j) * HID_DIM + col] = f2bf(v);
                }
            } else if (out_id == 2) {                // SiLU -> bf16, Vt[d][tok]
                u32 pk[4];
#pragma unroll
                for (int j = 0; j < 4; ++j) {
                    float v0 = silu_f(acc[sm][sn][2 * j]     + bc);
                    float v1 = silu_f(acc[sm][sn][2 * j + 1] + bc);
                    pk[j] = (u32)f2bf(v0) | ((u32)f2bf(v1) << 16);
                }
                uint4 st = make_uint4(pk[0], pk[1], pk[2], pk[3]);
                *(uint4*)(Vt + (size_t)col * T_TOK + row0) = st;
            } else {                                 // SiLU -> f32, row-major
#pragma unroll
                for (int j = 0; j < 8; ++j) {
                    float v = silu_f(acc[sm][sn][j] + bc);
                    Uf[(size_t)(row0 + j) * HID_DIM + col] = v;
                }
            }
        }
    }
}

// ---------------------------------------------------------------------------
// 2) Causal (masked, no-softmax) attention: O = (Q K^T .* tril) V per head.
//    One wave per (b, h, 16-row block). m processed in 32-wide chunks:
//    E (f32, C-layout) -> mask -> LDS re-swizzle -> bf16 A-fragment -> E@V.
// ---------------------------------------------------------------------------
__global__ __launch_bounds__(32)
void attn_kernel(const u16* __restrict__ Qh, const u16* __restrict__ Kh,
                 const u16* __restrict__ Vt, float* __restrict__ Of) {
    __shared__ u16 elds[16][40];           // 16x32 tile, padded row stride
    const int lane = threadIdx.x & 31;
    const int n0 = blockIdx.x << 4;        // row block start (0..2032)
    const int h  = blockIdx.y;
    const int b  = blockIdx.z;
    const int lr = lane & 15;
    const int rbase = (lane >> 4) << 3;
    const size_t tokbase = (size_t)b * N_SEQ;

    // Preload Q A-fragments for this 16-row block (whole D=128).
    const u16* qrow = Qh + (tokbase + n0 + lr) * HID_DIM + h * D_HEAD;
    v16bf qa[4];
#pragma unroll
    for (int kd = 0; kd < 4; ++kd) qa[kd] = load_frag(qrow + kd * 32, lane);

    v8f oacc[8] = {};

    for (int mc = 0; mc <= n0; mc += 32) {
        __syncthreads();                   // protect previous LDS loads
#pragma unroll
        for (int tt = 0; tt < 2; ++tt) {
            const int m0 = mc + tt * 16;
            const u16* krow = Kh + (tokbase + m0 + lr) * HID_DIM + h * D_HEAD;
            v8f e = {};
#pragma unroll
            for (int kd = 0; kd < 4; ++kd) {
                v16bf kb = load_frag(krow + kd * 32, lane);
                e = __builtin_amdgcn_wmma_f32_16x16x32_bf16(
                    false, qa[kd], false, kb, (short)0, e, false, false);
            }
            // mask (keep m <= n) and stash transposed as bf16
#pragma unroll
            for (int j = 0; j < 8; ++j) {
                const int i = rbase + j;           // local row (token n0+i)
                float v = e[j];
                if (m0 + lr > n0 + i) v = 0.0f;    // causal mask
                elds[i][tt * 16 + lr] = f2bf(v);
            }
        }
        __syncthreads();
        // Re-read as A-fragment (16x32, K = m-chunk)
        const int k0 = (lane & 16) ? 8 : 0;
        Frag16 ea;
        ea.u[0] = *(const uint4*)(&elds[lr][k0]);
        ea.u[1] = *(const uint4*)(&elds[lr][16 + k0]);
        // O += E @ V  (V is d-major: Vt[d][token] -> contiguous B loads)
#pragma unroll
        for (int dt = 0; dt < 8; ++dt) {
            const u16* vrow = Vt + (size_t)(h * D_HEAD + dt * 16 + lr) * T_TOK
                            + tokbase + mc;
            v16bf vb = load_frag(vrow, lane);
            oacc[dt] = __builtin_amdgcn_wmma_f32_16x16x32_bf16(
                false, ea.bf, false, vb, (short)0, oacc[dt], false, false);
        }
    }

#pragma unroll
    for (int dt = 0; dt < 8; ++dt) {
#pragma unroll
        for (int j = 0; j < 8; ++j) {
            const int row = n0 + rbase + j;
            const int col = h * D_HEAD + dt * 16 + lr;
            Of[(tokbase + row) * HID_DIM + col] = oacc[dt][j];
        }
    }
}

// ---------------------------------------------------------------------------
// 3) g = u * (scale * o / (rms + eps)) ; one block per token row.
// ---------------------------------------------------------------------------
__global__ __launch_bounds__(256)
void rmsnorm_mul_kernel(const float* __restrict__ Of,
                        const float* __restrict__ Uf,
                        const float* __restrict__ scale,
                        u16* __restrict__ Gh) {
    __shared__ float red[256];
    const int row = blockIdx.x;
    const int tid = threadIdx.x;
    const float* o = Of + (size_t)row * HID_DIM;
    float4 v = ((const float4*)o)[tid];
    red[tid] = v.x * v.x + v.y * v.y + v.z * v.z + v.w * v.w;
    __syncthreads();
    for (int s = 128; s > 0; s >>= 1) {
        if (tid < s) red[tid] += red[tid + s];
        __syncthreads();
    }
    const float rms = sqrtf(red[0] * (1.0f / (float)HID_DIM));
    const float inv = 1.0f / (rms + EPS_F);
    float4 u4 = ((const float4*)(Uf + (size_t)row * HID_DIM))[tid];
    float4 s4 = ((const float4*)scale)[tid];
    float g0 = s4.x * v.x * inv * u4.x;
    float g1 = s4.y * v.y * inv * u4.y;
    float g2 = s4.z * v.z * inv * u4.z;
    float g3 = s4.w * v.w * inv * u4.w;
    u32 lo = (u32)f2bf(g0) | ((u32)f2bf(g1) << 16);
    u32 hi = (u32)f2bf(g2) | ((u32)f2bf(g3) << 16);
    ((uint2*)(Gh + (size_t)row * HID_DIM))[tid] = make_uint2(lo, hi);
}

// ---------------------------------------------------------------------------
// 4) y = g @ Wo^T + bo  (f32 output). Same 32x64 wave tiling as proj.
// ---------------------------------------------------------------------------
__global__ __launch_bounds__(256)
void out_gemm_kernel(const u16* __restrict__ Gh, const u16* __restrict__ Woh,
                     const float* __restrict__ bo, float* __restrict__ Y) {
    const int lane = threadIdx.x & 31;
    const int wid  = (blockIdx.x * blockDim.x + threadIdx.x) >> 5;
    const int tm = (wid >> 4) << 5;
    const int tn = (wid & 15) << 6;
    const int lr = lane & 15;
    const int rbase = (lane >> 4) << 3;

    const u16* arow0 = Gh + (size_t)(tm + lr)      * HID_DIM;
    const u16* arow1 = Gh + (size_t)(tm + 16 + lr) * HID_DIM;
    const u16* brow[4];
#pragma unroll
    for (int s = 0; s < 4; ++s)
        brow[s] = Woh + (size_t)(tn + s * 16 + lr) * HID_DIM;

    v8f acc[2][4] = {};
    for (int kk = 0; kk < HID_DIM; kk += 32) {
        v16bf a0 = load_frag(arow0 + kk, lane);
        v16bf a1 = load_frag(arow1 + kk, lane);
#pragma unroll
        for (int s = 0; s < 4; ++s) {
            v16bf bfr = load_frag(brow[s] + kk, lane);
            acc[0][s] = __builtin_amdgcn_wmma_f32_16x16x32_bf16(
                false, a0, false, bfr, (short)0, acc[0][s], false, false);
            acc[1][s] = __builtin_amdgcn_wmma_f32_16x16x32_bf16(
                false, a1, false, bfr, (short)0, acc[1][s], false, false);
        }
    }

#pragma unroll
    for (int sm = 0; sm < 2; ++sm) {
#pragma unroll
        for (int sn = 0; sn < 4; ++sn) {
            const int col  = tn + sn * 16 + lr;
            const float bc = bo[col];
            const int row0 = tm + sm * 16 + rbase;
#pragma unroll
            for (int j = 0; j < 8; ++j)
                Y[(size_t)(row0 + j) * E_DIM + col] = acc[sm][sn][j] + bc;
        }
    }
}

// ---------------------------------------------------------------------------
// Host-side launcher
// ---------------------------------------------------------------------------
extern "C" void kernel_launch(void* const* d_in, const int* in_sizes, int n_in,
                              void* d_out, int out_size, void* d_ws, size_t ws_size,
                              hipStream_t stream) {
    (void)in_sizes; (void)n_in; (void)out_size; (void)ws_size;
    const float* x  = (const float*)d_in[0];
    const float* Wq = (const float*)d_in[1];
    const float* bq = (const float*)d_in[2];
    const float* Wk = (const float*)d_in[3];
    const float* bk = (const float*)d_in[4];
    const float* Wv = (const float*)d_in[5];
    const float* bv = (const float*)d_in[6];
    const float* Wu = (const float*)d_in[7];
    const float* bu = (const float*)d_in[8];
    const float* Wo = (const float*)d_in[9];
    const float* bo = (const float*)d_in[10];
    const float* sc = (const float*)d_in[11];
    float* Y = (float*)d_out;

    char* p = (char*)d_ws;
    u16*   xh  = (u16*)p;   p += (size_t)T_TOK * E_DIM * 2;          // 8 MB
    u16*   W4h = (u16*)p;   p += (size_t)4 * HID_DIM * E_DIM * 2;    // 8 MB
    u16*   Woh = (u16*)p;   p += (size_t)E_DIM * HID_DIM * 2;        // 2 MB
    u16*   Qh  = (u16*)p;   p += (size_t)T_TOK * HID_DIM * 2;        // 8 MB
    u16*   Kh  = (u16*)p;   p += (size_t)T_TOK * HID_DIM * 2;        // 8 MB
    u16*   Vt  = (u16*)p;   p += (size_t)HID_DIM * T_TOK * 2;        // 8 MB
    float* Uf  = (float*)p; p += (size_t)T_TOK * HID_DIM * 4;        // 16 MB
    float* Of  = (float*)p; p += (size_t)T_TOK * HID_DIM * 4;        // 16 MB
    u16*   Gh  = (u16*)p;   p += (size_t)T_TOK * HID_DIM * 2;        // 8 MB

    // 0) conversions to bf16
    {
        int n4 = (T_TOK * E_DIM) / 4;
        cvt_bf16_kernel<<<(n4 + 255) / 256, 256, 0, stream>>>(x, xh, n4);
        int w4 = (HID_DIM * E_DIM) / 4;
        int g  = (w4 + 255) / 256;
        cvt_bf16_kernel<<<g, 256, 0, stream>>>(Wq, W4h + 0 * (size_t)HID_DIM * E_DIM, w4);
        cvt_bf16_kernel<<<g, 256, 0, stream>>>(Wk, W4h + 1 * (size_t)HID_DIM * E_DIM, w4);
        cvt_bf16_kernel<<<g, 256, 0, stream>>>(Wv, W4h + 2 * (size_t)HID_DIM * E_DIM, w4);
        cvt_bf16_kernel<<<g, 256, 0, stream>>>(Wu, W4h + 3 * (size_t)HID_DIM * E_DIM, w4);
        cvt_bf16_kernel<<<g, 256, 0, stream>>>(Wo, Woh, w4);
    }

    // 1) fused QKVU projections: 4 outputs * 2048 tiles = 8192 waves
    proj_kernel<<<1024, 256, 0, stream>>>(xh, W4h, bq, bk, bv, bu, Qh, Kh, Vt, Uf);

    // 2) causal masked attention: one wave per (row-block, head, batch)
    attn_kernel<<<dim3(N_SEQ / 16, H_HEADS, 2), 32, 0, stream>>>(Qh, Kh, Vt, Of);

    // 3) RMSNorm * u -> bf16
    rmsnorm_mul_kernel<<<T_TOK, 256, 0, stream>>>(Of, Uf, sc, Gh);

    // 4) output projection: 2048 waves
    out_gemm_kernel<<<256, 256, 0, stream>>>(Gh, Woh, bo, Y);
}
